// GNN_63771674411478
// MI455X (gfx1250) — matrix operand order repaired and checked by reading, not compiled, not run on hardware
//
#include <hip/hip_runtime.h>
#include <hip/hip_bf16.h>

typedef __attribute__((ext_vector_type(16))) __bf16 v16bf;
typedef __attribute__((ext_vector_type(8)))  __bf16 v8bf;
typedef __attribute__((ext_vector_type(2)))  __bf16 v2bf;
typedef __attribute__((ext_vector_type(8)))  float  v8f;

#define N_NODES  50000
#define N_PAD    50048   // = 782*64 = 391*128 : covers both GEMM grids
#define NFEAT    512
#define HIDDEN   128
#define NC       40
#define NCP      48      // 40 padded to 3x16 tiles (support2 row stride)

// ---------------------------------------------------------------------------
// utility: zero fill
// ---------------------------------------------------------------------------
__global__ __launch_bounds__(256) void gcn_fill0(float* __restrict__ p, int n) {
    int i = blockIdx.x * 256 + threadIdx.x;
    if (i < n) p[i] = 0.0f;
}

// ---------------------------------------------------------------------------
// GEMM1: support[N_PAD,128] = x[N,512] @ W1[512,128]  (bf16 WMMA, f32 accum)
// block = 256 threads (8 waves), covers 64 rows x 128 cols.
// A rows CLAMPED on load; output rows PADDED -> completely branch-free.
// All 4 B fragments loaded first (one clause/wait), then 4 WMMAs issue
// back-to-back for XDL pipelining.
// ---------------------------------------------------------------------------
__global__ __launch_bounds__(256) void gcn_gemm1(const float* __restrict__ x,
                                                 const float* __restrict__ W1,
                                                 float* __restrict__ out) {
    __shared__ __bf16 As [64][32];      // 4 KB
    __shared__ __bf16 BsT[HIDDEN][32];  // 8 KB, TRANSPOSED: [col][k]

    const int tid  = threadIdx.x;
    const int lane = tid & 31;
    const int wave = tid >> 5;
    const int rowTile = wave & 3;
    const int colHalf = wave >> 2;
    const int rowBase = blockIdx.x * 64;

    // staging coords (loop-invariant)
    const int ar  = tid >> 2;                 // 0..63
    const int ak0 = (tid & 3) * 8;            // 0,8,16,24
    const int arow = rowBase + ar;
    const int asrow = (arow < N_NODES) ? arow : (N_NODES - 1);  // clamp
    const int bkp = tid >> 4;                 // 0..15 -> k = 2*bkp, 2*bkp+1
    const int bc0 = (tid & 15) * 8;           // 0..120

    v8f acc[4] = {};

    for (int kc = 0; kc < NFEAT / 32; ++kc) {
        // ---- stage A chunk: 2x global b128 -> packed bf16 -> 1x ds b128 ----
        {
            const float4* src = (const float4*)(x + (size_t)asrow * NFEAT + kc * 32 + ak0);
            float4 a0 = src[0], a1 = src[1];
            v8bf pk;
            pk[0] = (__bf16)a0.x; pk[1] = (__bf16)a0.y;
            pk[2] = (__bf16)a0.z; pk[3] = (__bf16)a0.w;
            pk[4] = (__bf16)a1.x; pk[5] = (__bf16)a1.y;
            pk[6] = (__bf16)a1.z; pk[7] = (__bf16)a1.w;
            *(v8bf*)&As[ar][ak0] = pk;
        }
        // ---- stage B chunk transposed: 2 K-rows/thread, packed b32 scatter ----
        {
            const float* r0 = W1 + (size_t)(kc * 32 + 2 * bkp) * HIDDEN + bc0;
            const float* r1 = r0 + HIDDEN;
            float4 q0 = ((const float4*)r0)[0], q1 = ((const float4*)r0)[1];
            float4 s0 = ((const float4*)r1)[0], s1 = ((const float4*)r1)[1];
            float lo[8] = {q0.x, q0.y, q0.z, q0.w, q1.x, q1.y, q1.z, q1.w};
            float hi[8] = {s0.x, s0.y, s0.z, s0.w, s1.x, s1.y, s1.z, s1.w};
            #pragma unroll
            for (int j = 0; j < 8; ++j) {
                v2bf pk;
                pk[0] = (__bf16)lo[j];
                pk[1] = (__bf16)hi[j];
                *(v2bf*)&BsT[bc0 + j][2 * bkp] = pk;
            }
        }
        __syncthreads();

        // ---- A fragment (contiguous k = 16*(lane/16) + s, same as B): 2x ds_load_b128 ----
        v16bf afrag;
        {
            int m  = lane & 15;
            int kb = (lane < 16) ? 0 : 16;
            #pragma unroll
            for (int j = 0; j < 8; ++j) {
                afrag[j]     = As[rowTile * 16 + m][kb + j];
                afrag[8 + j] = As[rowTile * 16 + m][kb + 8 + j];
            }
        }
        // ---- load ALL 4 B fragments (one clause, one wait) ----
        v16bf bfrag[4];
        {
            int n    = lane & 15;
            int koff = (lane < 16) ? 0 : 16;
            #pragma unroll
            for (int t = 0; t < 4; ++t) {
                int ct = colHalf * 4 + t;
                #pragma unroll
                for (int j = 0; j < 16; ++j)
                    bfrag[t][j] = BsT[ct * 16 + n][koff + j];
            }
        }
        // ---- 4 back-to-back WMMAs ----
        #pragma unroll
        for (int t = 0; t < 4; ++t)
            acc[t] = __builtin_amdgcn_wmma_f32_16x16x32_bf16(
                false, afrag, false, bfrag[t], (short)0, acc[t], false, false);
        __syncthreads();
    }

    // ---- store: single base, constant offsets, NO guards (padded rows) ----
    // C layout: VGPR v -> M = v + (lane>=16 ? 8:0), N = lane&15
    {
        int n   = lane & 15;
        int mhi = (lane >> 4) * 8;
        float* dst = out + (size_t)(rowBase + rowTile * 16 + mhi) * HIDDEN
                         + colHalf * 64 + n;
        #pragma unroll
        for (int t = 0; t < 4; ++t)
            #pragma unroll
            for (int v = 0; v < 8; ++v)
                dst[v * HIDDEN + t * 16] = acc[t][v];
    }
}

// ---------------------------------------------------------------------------
// SpMM layer 1: h[row] += val * support[col]   (F = 128, one wave per edge)
// support (25.6 MB) and h (25.6 MB) are L2-resident; atomics resolve in L2.
// ---------------------------------------------------------------------------
__global__ __launch_bounds__(256) void gcn_spmm1(const int* __restrict__ erow,
                                                 const int* __restrict__ ecol,
                                                 const float* __restrict__ ev,
                                                 const float* __restrict__ support,
                                                 float* __restrict__ h, int nEdges) {
    int e    = (blockIdx.x * 256 + threadIdx.x) >> 5;
    int lane = threadIdx.x & 31;
    if (e >= nEdges) return;
    int   r = erow[e];
    int   c = ecol[e];
    float v = ev[e];
    float4 m = *((const float4*)(support + (size_t)c * HIDDEN) + lane);
    float* dst = h + (size_t)r * HIDDEN + lane * 4;
    atomicAdd(dst + 0, m.x * v);
    atomicAdd(dst + 1, m.y * v);
    atomicAdd(dst + 2, m.z * v);
    atomicAdd(dst + 3, m.w * v);
}

// ---------------------------------------------------------------------------
// bias + relu + dropout(0.5) on h, in place. Deterministic hash mask.
// ---------------------------------------------------------------------------
__global__ __launch_bounds__(256) void gcn_bias_relu_drop(float* __restrict__ h,
                                                          const float* __restrict__ b1) {
    int i = blockIdx.x * 256 + threadIdx.x;
    if (i >= N_NODES * HIDDEN) return;
    int f = i & (HIDDEN - 1);
    float x = h[i] + b1[f];
    x = fmaxf(x, 0.0f);
    unsigned u = (unsigned)i * 2654435761u;
    u ^= u >> 16; u *= 2246822519u; u ^= u >> 13;
    h[i] = (u & 1u) ? (x * 2.0f) : 0.0f;   // keep-prob 0.5, scale 1/0.5
}

// ---------------------------------------------------------------------------
// GEMM2: support2[N_PAD,48] = h[N,128] @ W2[128,40]  (bf16 WMMA)
// block = 256 threads (8 waves), covers 128 rows; wave w owns row tile w,
// 3 col tiles. W2 cached whole in LDS, TRANSPOSED [col][k], zero padded.
// Output rows AND cols padded -> branch-free epilogue.
// ---------------------------------------------------------------------------
__global__ __launch_bounds__(256) void gcn_gemm2(const float* __restrict__ h,
                                                 const float* __restrict__ W2,
                                                 float* __restrict__ out) {
    __shared__ __bf16 As  [128][32];     // 8 KB
    __shared__ __bf16 BsT2[NCP][HIDDEN]; // 12 KB, transposed, zero padded

    const int tid  = threadIdx.x;
    const int lane = tid & 31;
    const int wave = tid >> 5;          // row tile 0..7
    const int rowBase = blockIdx.x * 128;

    // load W2 -> LDS transposed (bf16, padded) once
    for (int i = tid; i < HIDDEN * NCP; i += 256) {
        int k = i / NCP, c = i % NCP;
        float v = (c < NC) ? W2[k * NC + c] : 0.0f;
        BsT2[c][k] = (__bf16)v;
    }

    // staging coords
    const int ar  = tid >> 1;
    const int ak0 = (tid & 1) * 16;
    const int arow = rowBase + ar;
    const int asrow = (arow < N_NODES) ? arow : (N_NODES - 1);  // clamp

    v8f acc[3] = {};

    for (int kc = 0; kc < HIDDEN / 32; ++kc) {
        __syncthreads();  // protects As reuse (and BsT2 on first iter)
        // stage A: 4x global b128 -> packed bf16 -> 2x ds b128
        {
            const float4* src = (const float4*)(h + (size_t)asrow * HIDDEN + kc * 32 + ak0);
            float4 a0 = src[0], a1 = src[1], a2 = src[2], a3 = src[3];
            v8bf p0, p1;
            p0[0] = (__bf16)a0.x; p0[1] = (__bf16)a0.y;
            p0[2] = (__bf16)a0.z; p0[3] = (__bf16)a0.w;
            p0[4] = (__bf16)a1.x; p0[5] = (__bf16)a1.y;
            p0[6] = (__bf16)a1.z; p0[7] = (__bf16)a1.w;
            p1[0] = (__bf16)a2.x; p1[1] = (__bf16)a2.y;
            p1[2] = (__bf16)a2.z; p1[3] = (__bf16)a2.w;
            p1[4] = (__bf16)a3.x; p1[5] = (__bf16)a3.y;
            p1[6] = (__bf16)a3.z; p1[7] = (__bf16)a3.w;
            *(v8bf*)&As[ar][ak0]     = p0;
            *(v8bf*)&As[ar][ak0 + 8] = p1;
        }
        __syncthreads();

        v16bf afrag;
        {
            int m  = lane & 15;
            int kb = (lane < 16) ? 0 : 16;
            #pragma unroll
            for (int j = 0; j < 8; ++j) {
                afrag[j]     = As[wave * 16 + m][kb + j];
                afrag[8 + j] = As[wave * 16 + m][kb + 8 + j];
            }
        }
        // load all 3 B fragments, then 3 back-to-back WMMAs
        v16bf bfrag[3];
        {
            int n    = lane & 15;
            int koff = (lane < 16) ? 0 : 16;
            #pragma unroll
            for (int t = 0; t < 3; ++t)
                #pragma unroll
                for (int j = 0; j < 16; ++j)
                    bfrag[t][j] = BsT2[t * 16 + n][kc * 32 + koff + j];
        }
        #pragma unroll
        for (int t = 0; t < 3; ++t)
            acc[t] = __builtin_amdgcn_wmma_f32_16x16x32_bf16(
                false, afrag, false, bfrag[t], (short)0, acc[t], false, false);
    }

    // ---- store: single base, constant offsets, NO guards (rows+cols padded) ----
    {
        int n   = lane & 15;
        int mhi = (lane >> 4) * 8;
        float* dst = out + (size_t)(rowBase + wave * 16 + mhi) * NCP + n;
        #pragma unroll
        for (int t = 0; t < 3; ++t)
            #pragma unroll
            for (int v = 0; v < 8; ++v)
                dst[v * NCP + t * 16] = acc[t][v];
    }
}

// ---------------------------------------------------------------------------
// SpMM layer 2: out[row] += val * support2[col]  (support2 row stride = NCP)
// 8 lanes per edge x 5 feats.
// ---------------------------------------------------------------------------
__global__ __launch_bounds__(256) void gcn_spmm2(const int* __restrict__ erow,
                                                 const int* __restrict__ ecol,
                                                 const float* __restrict__ ev,
                                                 const float* __restrict__ support,
                                                 float* __restrict__ out, int nEdges) {
    int gid = blockIdx.x * 256 + threadIdx.x;
    int e   = gid >> 3;
    int sub = gid & 7;
    if (e >= nEdges) return;
    int   r = erow[e];
    int   c = ecol[e];
    float v = ev[e];
    const float* src = support + (size_t)c * NCP + sub * 5;
    float*       dst = out     + (size_t)r * NC  + sub * 5;
    #pragma unroll
    for (int j = 0; j < 5; ++j) atomicAdd(dst + j, src[j] * v);
}

// ---------------------------------------------------------------------------
// final bias + relu on d_out, in place
// ---------------------------------------------------------------------------
__global__ __launch_bounds__(256) void gcn_bias_relu2(float* __restrict__ out,
                                                      const float* __restrict__ b2) {
    int i = blockIdx.x * 256 + threadIdx.x;
    if (i >= N_NODES * NC) return;
    int f = i % NC;
    out[i] = fmaxf(out[i] + b2[f], 0.0f);
}

// ---------------------------------------------------------------------------
extern "C" void kernel_launch(void* const* d_in, const int* in_sizes, int n_in,
                              void* d_out, int out_size, void* d_ws, size_t ws_size,
                              hipStream_t stream) {
    const float* x    = (const float*)d_in[0];
    const int*   erow = (const int*)d_in[1];
    const int*   ecol = (const int*)d_in[2];
    const float* ev   = (const float*)d_in[3];
    const float* W1   = (const float*)d_in[4];
    const float* b1   = (const float*)d_in[5];
    const float* W2   = (const float*)d_in[6];
    const float* b2   = (const float*)d_in[7];
    float*       out  = (float*)d_out;
    const int nEdges  = in_sizes[1];

    // workspace: [support: N_PAD*128] [h: N*128]
    // support buffer reused for support2 [N_PAD*48] (fits: 2.4M < 6.4M floats)
    float* support = (float*)d_ws;
    float* h       = support + (size_t)N_PAD * HIDDEN;

    const int NH  = N_NODES * HIDDEN;  // 6.4M
    const int NO  = N_NODES * NC;      // 2.0M

    // zero accumulators (d_out/d_ws are poisoned; atomics need zeros)
    gcn_fill0<<<(NH + 255) / 256, 256, 0, stream>>>(h, NH);
    gcn_fill0<<<(NO + 255) / 256, 256, 0, stream>>>(out, NO);

    // layer 1
    gcn_gemm1<<<N_PAD / 64, 256, 0, stream>>>(x, W1, support);
    gcn_spmm1<<<((size_t)nEdges * 32 + 255) / 256, 256, 0, stream>>>(
        erow, ecol, ev, support, h, nEdges);
    gcn_bias_relu_drop<<<(NH + 255) / 256, 256, 0, stream>>>(h, b1);

    // layer 2 (reuse `support` buffer for padded [N_PAD,48])
    gcn_gemm2<<<N_PAD / 128, 256, 0, stream>>>(h, W2, support);
    gcn_spmm2<<<((size_t)nEdges * 8 + 255) / 256, 256, 0, stream>>>(
        erow, ecol, ev, support, out, nEdges);
    gcn_bias_relu2<<<(NO + 255) / 256, 256, 0, stream>>>(out, b2);
}
